// VectorQuantizer_90202903150916
// MI455X (gfx1250) — matrix operand order repaired and checked by reading, not compiled, not run on hardware
//
#include <hip/hip_runtime.h>

typedef __attribute__((ext_vector_type(16))) __bf16 v16bf;
typedef __attribute__((ext_vector_type(8)))  float  v8f;

#define VQ_D 256
#define VQ_K 1024
#define VQ_N 131072      // 8 * 16384 rows
#define TILES_K 64       // K / 16
#define CHUNKS 8         // D / 32
#define RED_BLOCKS 4096

// gfx1250 async global->LDS staging (ASYNCcnt-tracked, no VGPR round trip).
// Flip to 0 if the assembler rejects the mnemonic on this toolchain.
#define USE_ASYNC_LDS 1

// ---------------------------------------------------------------------------
// Kernel A: split codewords into bf16 hi/lo, compute c_sq[k] = ||c_k||^2
// ---------------------------------------------------------------------------
__global__ void vq_prep(const float* __restrict__ cw,
                        __bf16* __restrict__ cwHi, __bf16* __restrict__ cwLo,
                        float* __restrict__ cSq) {
  int k = blockIdx.x;       // 1024
  int d = threadIdx.x;      // 256
  float x = cw[k * VQ_D + d];
  __bf16 h = (__bf16)x;
  __bf16 l = (__bf16)(x - (float)h);
  cwHi[k * VQ_D + d] = h;
  cwLo[k * VQ_D + d] = l;
  float s = x * x;
  #pragma unroll
  for (int m = 16; m >= 1; m >>= 1) s += __shfl_xor(s, m, 32);
  __shared__ float red[8];
  if ((threadIdx.x & 31) == 0) red[threadIdx.x >> 5] = s;
  __syncthreads();
  if (threadIdx.x == 0) {
    float t = 0.f;
    #pragma unroll
    for (int w = 0; w < 8; ++w) t += red[w];
    cSq[k] = t;
  }
}

#if USE_ASYNC_LDS
// Copy 32 bytes: LDS[dst + {0,16}] = MEM[saddr + voff + {0,16}] (per lane).
__device__ __forceinline__ void async_copy32(unsigned dstLds, unsigned voff,
                                             const __bf16* __restrict__ src) {
  asm volatile("global_load_async_to_lds_b128 %0, %1, %2 offset:0"
               :: "v"(dstLds), "v"(voff), "s"(src) : "memory");
  asm volatile("global_load_async_to_lds_b128 %0, %1, %2 offset:16"
               :: "v"(dstLds), "v"(voff), "s"(src) : "memory");
}
__device__ __forceinline__ void wait_async0() {
  asm volatile("s_wait_asynccnt 0x0" ::: "memory");
}
#endif

// ---------------------------------------------------------------------------
// Kernel B: argmin_k (c_sq[k] - 2 x.c_k) via bf16 hi/lo WMMA.
// WG = 256 threads = 8 waves; each wave owns 16 input rows (A in VGPRs),
// codeword tiles (16 x 256 hi/lo bf16) double-buffered through LDS.
// ---------------------------------------------------------------------------
__launch_bounds__(256)
__global__ void vq_argmin(const float* __restrict__ x,
                          const __bf16* __restrict__ cwHi,
                          const __bf16* __restrict__ cwLo,
                          const float* __restrict__ cSq,
                          int* __restrict__ idxOut,
                          float* __restrict__ idxOutF) {
  __shared__ __attribute__((aligned(16))) __bf16 sB[2][2][16 * VQ_D]; // 32 KB
  const int tid  = threadIdx.x;
  const int wave = tid >> 5;
  const int lane = tid & 31;
  const int half = (lane >> 4) & 1;
  const int lrow = lane & 15;
  const int rowBase = blockIdx.x * 128 + wave * 16;

  // ---- A fragments: row (rowBase+lrow), bf16 16x32 layout per chunk ----
  v16bf Ah[CHUNKS], Al[CHUNKS];
  {
    const float* xr = x + (size_t)(rowBase + lrow) * VQ_D;
    #pragma unroll
    for (int c = 0; c < CHUNKS; ++c) {
      int k0 = c * 32 + half * 16;       // K = k0..k0+7   -> VGPR0..3
      int k1 = c * 32 + half * 16 + 8;   // K = k1..k1+7   -> VGPR4..7
      float4 f0 = *(const float4*)(xr + k0);
      float4 f1 = *(const float4*)(xr + k0 + 4);
      float4 f2 = *(const float4*)(xr + k1);
      float4 f3 = *(const float4*)(xr + k1 + 4);
      float v[16] = {f0.x, f0.y, f0.z, f0.w, f1.x, f1.y, f1.z, f1.w,
                     f2.x, f2.y, f2.z, f2.w, f3.x, f3.y, f3.z, f3.w};
      #pragma unroll
      for (int i = 0; i < 16; ++i) {
        __bf16 h = (__bf16)v[i];
        Ah[c][i] = h;
        Al[c][i] = (__bf16)(v[i] - (float)h);
      }
    }
  }

  float best[8];
  int   bidx[8];
  #pragma unroll
  for (int j = 0; j < 8; ++j) { best[j] = 3.0e38f; bidx[j] = 0; }

#if USE_ASYNC_LDS
  // LDS byte addresses: generic-pointer low 32 bits == LDS offset on gfx1250.
  const unsigned ldsBase = (unsigned)(uintptr_t)&sB[0][0][0];
  const unsigned voff    = (unsigned)tid * 32u;      // byte offset within tile
  // prologue: issue tile 0 into buf 0
  async_copy32(ldsBase + voff,        voff, cwHi);
  async_copy32(ldsBase + 8192 + voff, voff, cwLo);
#else
  uint4 pH0, pH1, pL0, pL1;
  {
    const uint4* sh = (const uint4*)cwHi;
    const uint4* sl = (const uint4*)cwLo;
    pH0 = sh[tid * 2]; pH1 = sh[tid * 2 + 1];
    pL0 = sl[tid * 2]; pL1 = sl[tid * 2 + 1];
  }
#endif

  for (int t = 0; t < TILES_K; ++t) {
    const int buf = t & 1;
#if USE_ASYNC_LDS
    wait_async0();          // this wave's copies for tile t are in LDS
    __syncthreads();        // everyone's copies done; prev compute done
    if (t + 1 < TILES_K) {  // issue tile t+1 into the other buffer
      const unsigned nb = ldsBase + (unsigned)(buf ^ 1) * 16384u;
      async_copy32(nb + voff,        voff, cwHi + (size_t)(t + 1) * 16 * VQ_D);
      async_copy32(nb + 8192 + voff, voff, cwLo + (size_t)(t + 1) * 16 * VQ_D);
    }
#else
    ((uint4*)sB[buf][0])[tid * 2]     = pH0;
    ((uint4*)sB[buf][0])[tid * 2 + 1] = pH1;
    ((uint4*)sB[buf][1])[tid * 2]     = pL0;
    ((uint4*)sB[buf][1])[tid * 2 + 1] = pL1;
    __syncthreads();
    if (t + 1 < TILES_K) {
      const uint4* sh = (const uint4*)(cwHi + (size_t)(t + 1) * 16 * VQ_D);
      const uint4* sl = (const uint4*)(cwLo + (size_t)(t + 1) * 16 * VQ_D);
      pH0 = sh[tid * 2]; pH1 = sh[tid * 2 + 1];
      pL0 = sl[tid * 2]; pL1 = sl[tid * 2 + 1];
    }
#endif
    const int n = t * 16 + lrow;          // this lane's codeword column
    const float csq = cSq[n];
    // three independent accumulation chains -> no intra-chunk WMMA RAW
    v8f ahh = {}; v8f ahl = {}; v8f alh = {};
    #pragma unroll
    for (int c = 0; c < CHUNKS; ++c) {
      const v16bf bh = *(const v16bf*)&sB[buf][0][lrow * VQ_D + c * 32 + half * 16];
      const v16bf bl = *(const v16bf*)&sB[buf][1][lrow * VQ_D + c * 32 + half * 16];
      ahh = __builtin_amdgcn_wmma_f32_16x16x32_bf16(false, Ah[c], false, bh,
                                                    (short)0, ahh, false, false);
      ahl = __builtin_amdgcn_wmma_f32_16x16x32_bf16(false, Ah[c], false, bl,
                                                    (short)0, ahl, false, false);
      alh = __builtin_amdgcn_wmma_f32_16x16x32_bf16(false, Al[c], false, bh,
                                                    (short)0, alh, false, false);
    }
    #pragma unroll
    for (int j = 0; j < 8; ++j) {
      float d = csq - 2.0f * (ahh[j] + ahl[j] + alh[j]);
      if (d < best[j]) { best[j] = d; bidx[j] = n; }   // strict < keeps first min
    }
#if !USE_ASYNC_LDS
    // (async path syncs at loop top)
#endif
  }

  // reduce over the 16 lanes of each half (columns), tie -> lower index
  #pragma unroll
  for (int j = 0; j < 8; ++j) {
    float bv = best[j];
    int   bi = bidx[j];
    #pragma unroll
    for (int m = 8; m >= 1; m >>= 1) {
      float ov = __shfl_xor(bv, m, 32);
      int   oi = __shfl_xor(bi, m, 32);
      if (ov < bv || (ov == bv && oi < bi)) { bv = ov; bi = oi; }
    }
    if (lrow == 0) {
      int r = rowBase + half * 8 + j;
      idxOut[r]  = bi;
      idxOutF[r] = (float)bi;
    }
  }
}

// ---------------------------------------------------------------------------
// Kernel C: gather quantized = codewords[idx], deterministic partial loss sums
// ---------------------------------------------------------------------------
__global__ void vq_gather(const float* __restrict__ x, const float* __restrict__ cw,
                          const int* __restrict__ idx, float* __restrict__ q,
                          float* __restrict__ partials) {
  const size_t total  = (size_t)VQ_N * VQ_D;
  const size_t stride = (size_t)gridDim.x * blockDim.x;
  float s = 0.f;
  for (size_t i = (size_t)blockIdx.x * blockDim.x + threadIdx.x; i < total; i += stride) {
    int n = (int)(i >> 8);
    int d = (int)(i & 255);
    float c  = cw[(size_t)idx[n] * VQ_D + d];
    float xv = x[i];
    q[i] = c;
    float df = xv - c;
    s += df * df;
  }
  #pragma unroll
  for (int m = 16; m >= 1; m >>= 1) s += __shfl_xor(s, m, 32);
  __shared__ float red[8];
  if ((threadIdx.x & 31) == 0) red[threadIdx.x >> 5] = s;
  __syncthreads();
  if (threadIdx.x == 0) {
    float t = 0.f;
    #pragma unroll
    for (int w = 0; w < 8; ++w) t += red[w];
    partials[blockIdx.x] = t;
  }
}

// ---------------------------------------------------------------------------
// Kernel D: finalize loss = 1.25 * mean((q - x)^2)
// ---------------------------------------------------------------------------
__global__ void vq_loss(const float* __restrict__ partials, float* __restrict__ lossOut) {
  float s = 0.f;
  for (int i = threadIdx.x; i < RED_BLOCKS; i += 256) s += partials[i];
  #pragma unroll
  for (int m = 16; m >= 1; m >>= 1) s += __shfl_xor(s, m, 32);
  __shared__ float red[8];
  if ((threadIdx.x & 31) == 0) red[threadIdx.x >> 5] = s;
  __syncthreads();
  if (threadIdx.x == 0) {
    float t = 0.f;
    #pragma unroll
    for (int w = 0; w < 8; ++w) t += red[w];
    lossOut[0] = 1.25f * t / (float)((size_t)VQ_N * VQ_D);
  }
}

// ---------------------------------------------------------------------------
extern "C" void kernel_launch(void* const* d_in, const int* in_sizes, int n_in,
                              void* d_out, int out_size, void* d_ws, size_t ws_size,
                              hipStream_t stream) {
  const float* x  = (const float*)d_in[0];   // (8,16384,256) fp32
  const float* cw = (const float*)d_in[1];   // (1024,256) fp32

  char* ws = (char*)d_ws;
  __bf16* cwHi    = (__bf16*)(ws);                                   // 512 KB
  __bf16* cwLo    = (__bf16*)(ws + 512 * 1024);                      // 512 KB
  float*  cSq     = (float*)(ws + 1024 * 1024);                      // 4 KB
  int*    idxI    = (int*)(ws + 1024 * 1024 + 4096);                 // 512 KB
  float*  partials= (float*)(ws + 1024 * 1024 + 4096 + 512 * 1024);  // 16 KB

  float* q       = (float*)d_out;                       // 33554432 floats
  float* idxF    = q + (size_t)VQ_N * VQ_D;             // 131072 floats
  float* lossOut = idxF + VQ_N;                         // 1 float

  vq_prep  <<<VQ_K,        256, 0, stream>>>(cw, cwHi, cwLo, cSq);
  vq_argmin<<<VQ_N / 128,  256, 0, stream>>>(x, cwHi, cwLo, cSq, idxI, idxF);
  vq_gather<<<RED_BLOCKS,  256, 0, stream>>>(x, cw, idxI, q, partials);
  vq_loss  <<<1,           256, 0, stream>>>(partials, lossOut);
}
